// OptimizedMultiHeadAttention_8839042695279
// MI455X (gfx1250) — matrix operand order repaired and checked by reading, not compiled, not run on hardware
//
#include <hip/hip_runtime.h>

// ---------------------------------------------------------------------------
// MI455X (gfx1250) fused multi-head attention forward.
// bf16 WMMA (v_wmma_f32_16x16x32_bf16) everywhere, fp32 accumulation,
// flash-attention online softmax, LDS staging for the P-fragment transpose.
// wave32 fragment layouts per CDNA5 ISA 7.12.2.
// This revision preloads all B/V fragments into distinct registers before the
// WMMA bursts so the scheduler can stagger s_wait_loadcnt instead of
// serializing load->wait(0)->wmma, and uses near-cache prefetch locality.
// ---------------------------------------------------------------------------

typedef __attribute__((ext_vector_type(16))) __bf16 bf16x16;
typedef __attribute__((ext_vector_type(8)))  __bf16 bf16x8;
typedef __attribute__((ext_vector_type(4)))  __bf16 bf16x4;
typedef __attribute__((ext_vector_type(8)))  float  f32x8;

namespace {
constexpr int kB  = 2;
constexpr int kS  = 2048;
constexpr int kE  = 1024;
constexpr int kH  = 16;
constexpr int kHD = 64;
constexpr int kM  = kB * kS;   // 4096 rows of x
constexpr int kN3 = 3 * kE;    // 3072 qkv outputs
}

#define DEVFN static __device__ __forceinline__

DEVFN f32x8 zero8() {
  f32x8 z;
#pragma unroll
  for (int i = 0; i < 8; ++i) z[i] = 0.0f;
  return z;
}

// A-fragment (16x32 bf16, M x K): lane = row (lane&15); lanes 0-15 hold
// K = [k0..k0+7] in v[0..3] and [k0+16..k0+23] in v[4..7]; lanes 16-31 hold
// K = [k0+8..k0+15] and [k0+24..k0+31].
DEVFN bf16x16 load_a_frag(const __bf16* __restrict__ base, int ld,
                          int row0, int k0, int lane) {
  const int r  = row0 + (lane & 15);
  const int kk = k0 + ((lane >> 4) << 3);
  const __bf16* p = base + (size_t)r * ld + kk;
  union { bf16x16 v; bf16x8 h[2]; } u;
  u.h[0] = *reinterpret_cast<const bf16x8*>(p);
  u.h[1] = *reinterpret_cast<const bf16x8*>(p + 16);
  return u.v;
}

// B-fragment (32x16 bf16, K x N): lane = column (lane&15); lanes 0-15 hold
// K = [k0..k0+7] and [k0+16..k0+23], lanes 16-31 hold K = [k0+8..k0+15] and [k0+24..k0+31] (2 per VGPR).
// Source: row n of a row-major [N,K] matrix (K-contiguous).
DEVFN bf16x16 load_b_frag(const __bf16* __restrict__ base, int ld,
                          int n0, int k0, int lane) {
  const int r  = n0 + (lane & 15);
  const int kk = k0 + ((lane >> 4) << 3);
  const __bf16* p = base + (size_t)r * ld + kk;
  union { bf16x16 v; bf16x8 h[2]; } u;
  u.h[0] = *reinterpret_cast<const bf16x8*>(p);
  u.h[1] = *reinterpret_cast<const bf16x8*>(p + 16);
  return u.v;
}

DEVFN f32x8 wmma_bf16(bf16x16 a, bf16x16 b, f32x8 c) {
  return __builtin_amdgcn_wmma_f32_16x16x32_bf16(
      /*neg_a=*/false, a, /*neg_b=*/false, b,
      /*c_mod=*/(short)0, c, /*reuse_a=*/false, /*reuse_b=*/false);
}

// ---------------------------------------------------------------------------
// fp32 -> bf16 conversion (vectorized 4-wide)
// ---------------------------------------------------------------------------
__global__ void mha_cvt_bf16(const float4* __restrict__ in,
                             __bf16* __restrict__ out, int n4) {
  int i = blockIdx.x * blockDim.x + threadIdx.x;
  if (i < n4) {
    float4 f = in[i];
    bf16x4 o;
    o[0] = (__bf16)f.x; o[1] = (__bf16)f.y;
    o[2] = (__bf16)f.z; o[3] = (__bf16)f.w;
    *reinterpret_cast<bf16x4*>(out + (size_t)i * 4) = o;
  }
}

// ---------------------------------------------------------------------------
// QKV projection: qkv[m,n] = x[m,:] . qkv_w[n,:] + qkv_b[n]
// Each wave computes a 32x64 tile. Q/K scattered to [B,H,S,HD] (bf16),
// V scattered transposed to [B,H,HD,S] so P@V B-fragments load contiguous.
// ---------------------------------------------------------------------------
__global__ void mha_qkv_gemm(const __bf16* __restrict__ xb,
                             const __bf16* __restrict__ wb,
                             const float* __restrict__ bias,
                             __bf16* __restrict__ qbuf,
                             __bf16* __restrict__ kbuf,
                             __bf16* __restrict__ vtbuf) {
  const int lane = threadIdx.x & 31;
  const int wave = threadIdx.x >> 5;
  const int task = blockIdx.x * 8 + wave;
  const int nTilesN = kN3 / 64;                 // 48
  if (task >= (kM / 32) * nTilesN) return;
  const int m0 = (task / nTilesN) * 32;
  const int n0 = (task % nTilesN) * 64;

  f32x8 acc[2][4];
#pragma unroll
  for (int mi = 0; mi < 2; ++mi)
#pragma unroll
    for (int j = 0; j < 4; ++j) acc[mi][j] = zero8();

  for (int kk = 0; kk < kE; kk += 32) {
    if (kk + 32 < kE) {
      __builtin_prefetch(xb + (size_t)(m0 + (lane & 15)) * kE + kk + 32, 0, 3);
      __builtin_prefetch(wb + (size_t)(n0 + (lane & 15)) * kE + kk + 32, 0, 3);
    }
    // Issue all 12 b128 loads before any WMMA so waits can be staggered.
    bf16x16 a0 = load_a_frag(xb, kE, m0, kk, lane);
    bf16x16 a1 = load_a_frag(xb, kE, m0 + 16, kk, lane);
    bf16x16 bf[4];
#pragma unroll
    for (int j = 0; j < 4; ++j)
      bf[j] = load_b_frag(wb, kE, n0 + j * 16, kk, lane);
#pragma unroll
    for (int j = 0; j < 4; ++j) {
      acc[0][j] = wmma_bf16(a0, bf[j], acc[0][j]);
      acc[1][j] = wmma_bf16(a1, bf[j], acc[1][j]);
    }
  }

  const int colL = lane & 15;
  const int rofs = (lane >> 4) << 3;
#pragma unroll
  for (int j = 0; j < 4; ++j) {
    const int n = n0 + j * 16 + colL;
    const float bv = bias[n];
    const int which = n >> 10;         // 0=q 1=k 2=v (uniform per tile)
    const int hh = (n & (kE - 1)) >> 6;
    const int d = n & (kHD - 1);
#pragma unroll
    for (int mi = 0; mi < 2; ++mi) {
#pragma unroll
      for (int i = 0; i < 8; ++i) {
        const int row = m0 + mi * 16 + i + rofs;
        const int bb = row >> 11;            // row / S
        const int ss = row & (kS - 1);
        const __bf16 e = (__bf16)(acc[mi][j][i] + bv);
        if (which == 0)
          qbuf[(((size_t)(bb * kH + hh)) * kS + ss) * kHD + d] = e;
        else if (which == 1)
          kbuf[(((size_t)(bb * kH + hh)) * kS + ss) * kHD + d] = e;
        else
          vtbuf[(((size_t)(bb * kH + hh)) * kHD + d) * kS + ss] = e;
      }
    }
  }
}

// ---------------------------------------------------------------------------
// Flash attention: one wave per (b, h, 16-row q tile). Streams keys in
// 32-chunks: 4 WMMAs for scores, online softmax, P transposed through a
// per-wave 1KB LDS buffer into A-fragment layout, 4 WMMAs for P@V.
// ---------------------------------------------------------------------------
__global__ void mha_flash_attn(const __bf16* __restrict__ Q,
                               const __bf16* __restrict__ K,
                               const __bf16* __restrict__ VT,
                               __bf16* __restrict__ CTX) {
  __shared__ __align__(16) __bf16 pbuf[8][16 * 32];
  const int lane = threadIdx.x & 31;
  const int wave = threadIdx.x >> 5;
  const int task = blockIdx.x * 8 + wave;
  if (task >= kB * kH * (kS / 16)) return;
  const int qt = task & (kS / 16 - 1);
  const int h = (task >> 7) & (kH - 1);
  const int b = task >> 11;
  const int q0 = qt * 16;

  const __bf16* Qh = Q + ((size_t)(b * kH + h)) * kS * kHD;
  const __bf16* Kh = K + ((size_t)(b * kH + h)) * kS * kHD;
  const __bf16* Vh = VT + ((size_t)(b * kH + h)) * kHD * kS;

  const bf16x16 qa0 = load_a_frag(Qh, kHD, q0, 0, lane);
  const bf16x16 qa1 = load_a_frag(Qh, kHD, q0, 32, lane);

  f32x8 o[4];
#pragma unroll
  for (int t = 0; t < 4; ++t) o[t] = zero8();
  float m[8], l[8];
#pragma unroll
  for (int i = 0; i < 8; ++i) { m[i] = -1e30f; l[i] = 0.0f; }

  const int colL = lane & 15;
  const int rofs = (lane >> 4) << 3;
  __bf16* pw = &pbuf[wave][0];
  const float sc = 0.125f;  // 1/sqrt(64)

  for (int kb = 0; kb < q0 + 16; kb += 32) {
    // Preload all 4 K fragments, then 4 score WMMAs.
    bf16x16 b00 = load_b_frag(Kh, kHD, kb, 0, lane);
    bf16x16 b01 = load_b_frag(Kh, kHD, kb, 32, lane);
    bf16x16 b10 = load_b_frag(Kh, kHD, kb + 16, 0, lane);
    bf16x16 b11 = load_b_frag(Kh, kHD, kb + 16, 32, lane);
    f32x8 s0 = zero8(), s1 = zero8();
    s0 = wmma_bf16(qa0, b00, s0);
    s0 = wmma_bf16(qa1, b01, s0);
    s1 = wmma_bf16(qa0, b10, s1);
    s1 = wmma_bf16(qa1, b11, s1);

    // Preload V fragments early; consumed after the softmax update below.
    bf16x16 vf[4];
#pragma unroll
    for (int t = 0; t < 4; ++t)
      vf[t] = load_b_frag(Vh, kS, t * 16, kb, lane);

    const bool needMask = (kb + 31 > q0);
    float mx[8], al[8], rs[8];
#pragma unroll
    for (int i = 0; i < 8; ++i) {
      const int row = q0 + i + rofs;
      float v0 = s0[i] * sc;
      float v1 = s1[i] * sc;
      if (needMask) {
        if (kb + colL > row) v0 = -1e30f;
        if (kb + 16 + colL > row) v1 = -1e30f;
      }
      s0[i] = v0; s1[i] = v1;
      mx[i] = fmaxf(v0, v1);
    }
#pragma unroll
    for (int i = 0; i < 8; ++i) {
#pragma unroll
      for (int off = 8; off > 0; off >>= 1)
        mx[i] = fmaxf(mx[i], __shfl_xor(mx[i], off, 16));
    }
#pragma unroll
    for (int i = 0; i < 8; ++i) {
      const float mn = fmaxf(m[i], mx[i]);
      al[i] = __expf(m[i] - mn);
      m[i] = mn;
      const float p0 = __expf(s0[i] - mn);
      const float p1 = __expf(s1[i] - mn);
      s0[i] = p0; s1[i] = p1;
      rs[i] = p0 + p1;
    }
#pragma unroll
    for (int i = 0; i < 8; ++i) {
#pragma unroll
      for (int off = 8; off > 0; off >>= 1)
        rs[i] += __shfl_xor(rs[i], off, 16);
      l[i] = l[i] * al[i] + rs[i];
    }
#pragma unroll
    for (int t = 0; t < 4; ++t)
#pragma unroll
      for (int i = 0; i < 8; ++i) o[t][i] *= al[i];

    // P (C-layout) -> LDS -> reload in A-fragment layout.
#pragma unroll
    for (int i = 0; i < 8; ++i) {
      const int r = i + rofs;
      pw[r * 32 + colL] = (__bf16)s0[i];
      pw[r * 32 + 16 + colL] = (__bf16)s1[i];
    }
    asm volatile("s_wait_dscnt 0x0" ::: "memory");
    union { bf16x16 v; bf16x8 h2[2]; } up;
    const __bf16* pr = pw + (size_t)colL * 32 + rofs;
    up.h2[0] = *reinterpret_cast<const bf16x8*>(pr);
    up.h2[1] = *reinterpret_cast<const bf16x8*>(pr + 16);
    const bf16x16 pf = up.v;

#pragma unroll
    for (int t = 0; t < 4; ++t)
      o[t] = wmma_bf16(pf, vf[t], o[t]);
  }

  float inv[8];
#pragma unroll
  for (int i = 0; i < 8; ++i) inv[i] = 1.0f / l[i];
#pragma unroll
  for (int t = 0; t < 4; ++t) {
#pragma unroll
    for (int i = 0; i < 8; ++i) {
      const int row = q0 + i + rofs;
      const int col = h * kHD + t * 16 + colL;
      CTX[((size_t)(b * kS + row)) * kE + col] = (__bf16)(o[t][i] * inv[i]);
    }
  }
}

// ---------------------------------------------------------------------------
// Output projection: out[m,n] = ctx[m,:] . out_w[n,:] + out_b[n]  (fp32 out)
// ---------------------------------------------------------------------------
__global__ void mha_out_gemm(const __bf16* __restrict__ ctx,
                             const __bf16* __restrict__ wb,
                             const float* __restrict__ bias,
                             float* __restrict__ out) {
  const int lane = threadIdx.x & 31;
  const int wave = threadIdx.x >> 5;
  const int task = blockIdx.x * 8 + wave;
  const int nTilesN = kE / 64;                  // 16
  if (task >= (kM / 32) * nTilesN) return;
  const int m0 = (task / nTilesN) * 32;
  const int n0 = (task % nTilesN) * 64;

  f32x8 acc[2][4];
#pragma unroll
  for (int mi = 0; mi < 2; ++mi)
#pragma unroll
    for (int j = 0; j < 4; ++j) acc[mi][j] = zero8();

  for (int kk = 0; kk < kE; kk += 32) {
    if (kk + 32 < kE) {
      __builtin_prefetch(ctx + (size_t)(m0 + (lane & 15)) * kE + kk + 32, 0, 3);
      __builtin_prefetch(wb + (size_t)(n0 + (lane & 15)) * kE + kk + 32, 0, 3);
    }
    bf16x16 a0 = load_a_frag(ctx, kE, m0, kk, lane);
    bf16x16 a1 = load_a_frag(ctx, kE, m0 + 16, kk, lane);
    bf16x16 bf[4];
#pragma unroll
    for (int j = 0; j < 4; ++j)
      bf[j] = load_b_frag(wb, kE, n0 + j * 16, kk, lane);
#pragma unroll
    for (int j = 0; j < 4; ++j) {
      acc[0][j] = wmma_bf16(a0, bf[j], acc[0][j]);
      acc[1][j] = wmma_bf16(a1, bf[j], acc[1][j]);
    }
  }

  const int colL = lane & 15;
  const int rofs = (lane >> 4) << 3;
#pragma unroll
  for (int j = 0; j < 4; ++j) {
    const int n = n0 + j * 16 + colL;
    const float bv = bias[n];
#pragma unroll
    for (int mi = 0; mi < 2; ++mi) {
#pragma unroll
      for (int i = 0; i < 8; ++i) {
        const int row = m0 + mi * 16 + i + rofs;
        out[(size_t)row * kE + n] = acc[mi][j][i] + bv;
      }
    }
  }
}

// ---------------------------------------------------------------------------
extern "C" void kernel_launch(void* const* d_in, const int* in_sizes, int n_in,
                              void* d_out, int out_size, void* d_ws, size_t ws_size,
                              hipStream_t stream) {
  (void)in_sizes; (void)n_in; (void)out_size; (void)ws_size;
  const float* x    = (const float*)d_in[0];  // [B,S,E]
  const float* qkvw = (const float*)d_in[1];  // [3E,E]
  const float* qkvb = (const float*)d_in[2];  // [3E]
  const float* outw = (const float*)d_in[3];  // [E,E]
  const float* outb = (const float*)d_in[4];  // [E]
  float* out = (float*)d_out;                 // [B,S,E]

  // Workspace layout (bf16, 48 MB total)
  char* ws = (char*)d_ws;
  const size_t MB = 1024 * 1024;
  __bf16* xb   = (__bf16*)(ws + 0 * MB);    // [4096,1024]   8 MB
  __bf16* wqkv = (__bf16*)(ws + 8 * MB);    // [3072,1024]   6 MB
  __bf16* wout = (__bf16*)(ws + 14 * MB);   // [1024,1024]   2 MB
  __bf16* qb   = (__bf16*)(ws + 16 * MB);   // [B,H,S,HD]    8 MB
  __bf16* kb   = (__bf16*)(ws + 24 * MB);   // [B,H,S,HD]    8 MB
  __bf16* vtb  = (__bf16*)(ws + 32 * MB);   // [B,H,HD,S]    8 MB
  __bf16* ctx  = (__bf16*)(ws + 40 * MB);   // [B,S,E]       8 MB

  // 1) fp32 -> bf16 conversions
  {
    int n4 = kM * kE / 4;       // 1,048,576
    mha_cvt_bf16<<<(n4 + 255) / 256, 256, 0, stream>>>((const float4*)x, xb, n4);
    n4 = kN3 * kE / 4;          // 786,432
    mha_cvt_bf16<<<(n4 + 255) / 256, 256, 0, stream>>>((const float4*)qkvw, wqkv, n4);
    n4 = kE * kE / 4;           // 262,144
    mha_cvt_bf16<<<(n4 + 255) / 256, 256, 0, stream>>>((const float4*)outw, wout, n4);
  }

  // 2) QKV projection (6144 wave-tasks, 8 waves/block)
  mha_qkv_gemm<<<(kM / 32) * (kN3 / 64) / 8, 256, 0, stream>>>(
      xb, wqkv, qkvb, qb, kb, vtb);

  // 3) Flash attention (4096 wave-tasks)
  mha_flash_attn<<<kB * kH * (kS / 16) / 8, 256, 0, stream>>>(qb, kb, vtb, ctx);

  // 4) Output projection (2048 wave-tasks)
  mha_out_gemm<<<(kM / 32) * (kE / 64) / 8, 256, 0, stream>>>(ctx, wout, outb, out);
}
